// OUPlayerNetwork_57775900065922
// MI455X (gfx1250) — matrix operand order repaired and checked by reading, not compiled, not run on hardware
//
#include <hip/hip_runtime.h>

// ---------------- CDNA5 WMMA types ----------------
typedef __bf16 v16bf __attribute__((ext_vector_type(16)));
typedef float  v8f   __attribute__((ext_vector_type(8)));

// ---------------- model dims ----------------
#define B_   2048
#define T_   6
#define T2_  12
#define P_   512
#define H_   1024
#define NH_  8
#define HD_  128
#define A_   13
#define F_   128
#define BT_  (B_*T_)     // 12288
#define B12_ (B_*T2_)    // 24576

// ================= bf16 fragment packing =================
__device__ __forceinline__ v16bf pack16(float4 x0, float4 x1, float4 x2, float4 x3) {
  v16bf r;
  r[0]=(__bf16)x0.x;  r[1]=(__bf16)x0.y;  r[2]=(__bf16)x0.z;  r[3]=(__bf16)x0.w;
  r[4]=(__bf16)x1.x;  r[5]=(__bf16)x1.y;  r[6]=(__bf16)x1.z;  r[7]=(__bf16)x1.w;
  r[8]=(__bf16)x2.x;  r[9]=(__bf16)x2.y;  r[10]=(__bf16)x2.z; r[11]=(__bf16)x2.w;
  r[12]=(__bf16)x3.x; r[13]=(__bf16)x3.y; r[14]=(__bf16)x3.z; r[15]=(__bf16)x3.w;
  return r;
}
// A-matrix fragment: lane covers 8 floats at p[0..7] and 8 at p[16..23]
__device__ __forceinline__ v16bf load_a_frag(const float* __restrict__ p) {
  const float4* q0 = reinterpret_cast<const float4*>(p);
  const float4* q1 = reinterpret_cast<const float4*>(p + 16);
  return pack16(q0[0], q0[1], q1[0], q1[1]);
}
// B-matrix fragment: 16 contiguous floats
__device__ __forceinline__ v16bf load_b_frag(const float* __restrict__ p) {
  const float4* q = reinterpret_cast<const float4*>(p);
  return pack16(q[0], q[1], q[2], q[3]);
}

// ================= WMMA GEMM =================
// Y[m,n] = act( sum_k X[m,k]*W[n,k] + bias[n] (+ residual[ym,n]) )
// X:[M,K] row-major, W:[N,K] row-major (PyTorch layout; computes X @ W^T).
// One wave per 64x32 C tile (4x2 of 16x16 WMMA): 8 WMMAs per K-step with
// 4 A-frag + 2 B-frag loads => 3 b128 loads per WMMA, 8 independent acc chains.
// Requires M%64==0, N%32==0, K%32==0.
// rowmap: 0 identity; 1: ym = (m/6)*12 + m%6  (scatter T=6 rows into [B,12,H]).
__global__ __launch_bounds__(32)
void gemm_wmma_k(const float* __restrict__ X, const float* __restrict__ W,
                 const float* __restrict__ bias, const float* __restrict__ residual,
                 float* __restrict__ Y, int M, int N, int K, int act, int rowmap)
{
  const int lane = threadIdx.x;
  const int half = lane >> 4;     // 0/1
  const int l16  = lane & 15;
  const int n0 = blockIdx.x * 32;
  const int m0 = blockIdx.y * 64;

  const float* Ar[4];
  #pragma unroll
  for (int i = 0; i < 4; ++i)
    Ar[i] = X + (size_t)(m0 + i * 16 + l16) * K + half * 16;
  const float* Br[2];
  #pragma unroll
  for (int j = 0; j < 2; ++j)
    Br[j] = W + (size_t)(n0 + j * 16 + l16) * K + half * 16;

  v8f acc[4][2];
  #pragma unroll
  for (int i = 0; i < 4; ++i)
    #pragma unroll
    for (int j = 0; j < 2; ++j)
      #pragma unroll
      for (int e = 0; e < 8; ++e) acc[i][j][e] = 0.0f;

  for (int kt = 0; kt < K; kt += 32) {
    __builtin_prefetch(Ar[0] + kt + 64, 0, 1);
    __builtin_prefetch(Br[0] + kt + 64, 0, 1);
    v16bf a[4], bf[2];
    #pragma unroll
    for (int i = 0; i < 4; ++i) a[i] = load_b_frag(Ar[i] + kt);
    #pragma unroll
    for (int j = 0; j < 2; ++j) bf[j] = load_b_frag(Br[j] + kt);
    #pragma unroll
    for (int i = 0; i < 4; ++i) {
      #pragma unroll
      for (int j = 0; j < 2; ++j) {
        acc[i][j] = __builtin_amdgcn_wmma_f32_16x16x32_bf16(
            false, a[i], false, bf[j], (short)0, acc[i][j], false, false);
      }
    }
  }

  // C/D layout: VGPR r -> M = r + half*8 (+16 per m-tile), N = l16 (+16 per n-tile)
  #pragma unroll
  for (int i = 0; i < 4; ++i) {
    #pragma unroll
    for (int j = 0; j < 2; ++j) {
      const int n = n0 + j * 16 + l16;
      const float bsv = bias ? bias[n] : 0.0f;
      #pragma unroll
      for (int r = 0; r < 8; ++r) {
        int m = m0 + i * 16 + half * 8 + r;
        int ym = m;
        if (rowmap == 1) ym = (m / 6) * 12 + (m % 6);
        float v = acc[i][j][r] + bsv;
        if (residual) v += residual[(size_t)ym * N + n];
        if (act == 1) v = v > 0.0f ? v : 0.0f;
        Y[(size_t)ym * N + n] = v;
      }
    }
  }
}

// ================= attention (tiny seqlen, VALU) =================
// one wave per (b, h, q); qkv packed [B*S, 3H]; ctx [B*S, H]
__global__ __launch_bounds__(128)
void attn_self_k(const float* __restrict__ qkv, float* __restrict__ ctx,
                 int Bn, int S, float scale)
{
  const int wid  = blockIdx.x * (blockDim.x >> 5) + (threadIdx.x >> 5);
  const int lane = threadIdx.x & 31;
  const int total = Bn * NH_ * S;
  if (wid >= total) return;
  const int q = wid % S;
  const int bh = wid / S;
  const int h = bh % NH_;
  const int b = bh / NH_;
  const size_t row3 = (size_t)(b * S + q) * (3 * H_);
  const float* qptr = qkv + row3 + h * HD_;
  float qv[4];
  #pragma unroll
  for (int d = 0; d < 4; ++d) qv[d] = qptr[lane * 4 + d];

  float sc[12]; float mx = -1e30f;
  for (int k = 0; k < S; ++k) {
    const float* kp = qkv + (size_t)(b * S + k) * (3 * H_) + H_ + h * HD_;
    float p = 0.0f;
    #pragma unroll
    for (int d = 0; d < 4; ++d) p += qv[d] * kp[lane * 4 + d];
    #pragma unroll
    for (int off = 16; off > 0; off >>= 1) p += __shfl_xor(p, off, 32);
    p *= scale;
    sc[k] = p; mx = fmaxf(mx, p);
  }
  float den = 0.0f;
  for (int k = 0; k < S; ++k) { sc[k] = __expf(sc[k] - mx); den += sc[k]; }
  const float inv = 1.0f / den;
  float o[4] = {0, 0, 0, 0};
  for (int k = 0; k < S; ++k) {
    const float* vp = qkv + (size_t)(b * S + k) * (3 * H_) + 2 * H_ + h * HD_;
    const float w = sc[k] * inv;
    #pragma unroll
    for (int d = 0; d < 4; ++d) o[d] += w * vp[lane * 4 + d];
  }
  float* op = ctx + (size_t)(b * S + q) * H_ + h * HD_;
  #pragma unroll
  for (int d = 0; d < 4; ++d) op[lane * 4 + d] = o[d];
}

// one wave per (b, h); q [B,H]; kv [B*S, 2H] (k||v); ctx [B,H]
__global__ __launch_bounds__(128)
void attn_cross_k(const float* __restrict__ qp, const float* __restrict__ kv,
                  float* __restrict__ ctx, int Bn, int S, float scale)
{
  const int wid  = blockIdx.x * (blockDim.x >> 5) + (threadIdx.x >> 5);
  const int lane = threadIdx.x & 31;
  if (wid >= Bn * NH_) return;
  const int h = wid % NH_;
  const int b = wid / NH_;
  const float* qptr = qp + (size_t)b * H_ + h * HD_;
  float qv[4];
  #pragma unroll
  for (int d = 0; d < 4; ++d) qv[d] = qptr[lane * 4 + d];

  float sc[12]; float mx = -1e30f;
  for (int k = 0; k < S; ++k) {
    const float* kp = kv + (size_t)(b * S + k) * (2 * H_) + h * HD_;
    float p = 0.0f;
    #pragma unroll
    for (int d = 0; d < 4; ++d) p += qv[d] * kp[lane * 4 + d];
    #pragma unroll
    for (int off = 16; off > 0; off >>= 1) p += __shfl_xor(p, off, 32);
    p *= scale;
    sc[k] = p; mx = fmaxf(mx, p);
  }
  float den = 0.0f;
  for (int k = 0; k < S; ++k) { sc[k] = __expf(sc[k] - mx); den += sc[k]; }
  const float inv = 1.0f / den;
  float o[4] = {0, 0, 0, 0};
  for (int k = 0; k < S; ++k) {
    const float* vp = kv + (size_t)(b * S + k) * (2 * H_) + H_ + h * HD_;
    const float w = sc[k] * inv;
    #pragma unroll
    for (int d = 0; d < 4; ++d) o[d] += w * vp[lane * 4 + d];
  }
  float* op = ctx + (size_t)b * H_ + h * HD_;
  #pragma unroll
  for (int d = 0; d < 4; ++d) op[lane * 4 + d] = o[d];
}

// ================= layernorm: one wave per row =================
__global__ __launch_bounds__(128)
void layernorm_k(const float* __restrict__ x, const float* __restrict__ g,
                 const float* __restrict__ b, float* __restrict__ y, int rows)
{
  const int wid  = blockIdx.x * (blockDim.x >> 5) + (threadIdx.x >> 5);
  const int lane = threadIdx.x & 31;
  if (wid >= rows) return;
  const float* xr = x + (size_t)wid * H_;
  float s = 0.0f, s2 = 0.0f;
  for (int i = lane; i < H_; i += 32) { float v = xr[i]; s += v; s2 += v * v; }
  #pragma unroll
  for (int off = 16; off > 0; off >>= 1) { s += __shfl_xor(s, off, 32); s2 += __shfl_xor(s2, off, 32); }
  const float m = s * (1.0f / H_);
  const float var = s2 * (1.0f / H_) - m * m;
  const float inv = rsqrtf(var + 1e-5f);
  float* yr = y + (size_t)wid * H_;
  for (int i = lane; i < H_; i += 32) yr[i] = (xr[i] - m) * inv * g[i] + b[i];
}

// ================= elementwise helpers =================
__global__ void blend_k(const float* __restrict__ rev, const float* __restrict__ pred,
                        const int* __restrict__ mask, float* __restrict__ out)
{
  size_t i = (size_t)blockIdx.x * blockDim.x + threadIdx.x;
  if (i >= (size_t)BT_ * P_) return;
  const int bt = (int)(i / P_);
  const float m = (float)mask[bt];
  out[i] = rev[i] * m + pred[i] * (1.0f - m);
}

__global__ void copy_opp_k(const float* __restrict__ opp, float* __restrict__ x)
{
  size_t i = (size_t)blockIdx.x * blockDim.x + threadIdx.x;
  if (i >= (size_t)BT_ * H_) return;
  const int h = (int)(i % H_);
  const int t = (int)((i / H_) % T_);
  const int b = (int)(i / ((size_t)T_ * H_));
  x[((size_t)b * 12 + 6 + t) * H_ + h] = opp[i];
}

__global__ void gather_k(const float* __restrict__ x, const int* __restrict__ idx,
                         float* __restrict__ out)
{
  size_t i = (size_t)blockIdx.x * blockDim.x + threadIdx.x;
  if (i >= (size_t)B_ * H_) return;
  const int b = (int)(i / H_);
  const int h = (int)(i % H_);
  out[i] = x[((size_t)b * 12 + idx[b]) * H_ + h];
}

__global__ void mean12_k(const float* __restrict__ x, float* __restrict__ out)
{
  size_t i = (size_t)blockIdx.x * blockDim.x + threadIdx.x;
  if (i >= (size_t)B_ * H_) return;
  const int b = (int)(i / H_);
  const int h = (int)(i % H_);
  float s = 0.0f;
  #pragma unroll
  for (int t = 0; t < 12; ++t) s += x[((size_t)b * 12 + t) * H_ + h];
  out[i] = s * (1.0f / 12.0f);
}

__global__ void feats_k(const float* __restrict__ g, const float* __restrict__ cr,
                        const float* __restrict__ fh, const float* __restrict__ tera_table,
                        const int* __restrict__ tera_used, float* __restrict__ feats)
{
  const int FT = 3 * H_ + H_ / 4;
  size_t i = (size_t)blockIdx.x * blockDim.x + threadIdx.x;
  if (i >= (size_t)B_ * FT) return;
  const int b = (int)(i / FT);
  const int c = (int)(i % FT);
  float v;
  if (c < H_)            v = g[(size_t)b * H_ + c];
  else if (c < 2 * H_)   v = cr[(size_t)b * H_ + (c - H_)];
  else if (c < 3 * H_)   v = fh[(size_t)b * H_ + (c - 2 * H_)];
  else {
    const int idx = (tera_used[0] != 0) ? 0 : 1;   // ref: 0 if tera_used else 1
    v = tera_table[idx * (H_ / 4) + (c - 3 * H_)];
  }
  feats[i] = v;
}

// small-N head: one wave per (m,n)
__global__ __launch_bounds__(128)
void head_k(const float* __restrict__ X, const float* __restrict__ W,
            const float* __restrict__ bias, float* __restrict__ Y, int M, int N, int K)
{
  const int wid  = blockIdx.x * (blockDim.x >> 5) + (threadIdx.x >> 5);
  const int lane = threadIdx.x & 31;
  if (wid >= M * N) return;
  const int n = wid % N;
  const int m = wid / N;
  const float* xr = X + (size_t)m * K;
  const float* wr = W + (size_t)n * K;
  float s = 0.0f;
  for (int k = lane; k < K; k += 32) s += xr[k] * wr[k];
  #pragma unroll
  for (int off = 16; off > 0; off >>= 1) s += __shfl_xor(s, off, 32);
  if (lane == 0) Y[(size_t)m * N + n] = s + bias[n];
}

__global__ void policy_k(const float* __restrict__ logits, const int* __restrict__ mask,
                         float* __restrict__ pol)
{
  const int b = blockIdx.x * blockDim.x + threadIdx.x;
  if (b >= B_) return;
  float mx = -1e30f;
  for (int a = 0; a < A_; ++a)
    if (mask[b * A_ + a] != 0) mx = fmaxf(mx, logits[b * A_ + a]);
  float den = 0.0f;
  for (int a = 0; a < A_; ++a)
    if (mask[b * A_ + a] != 0) den += __expf(logits[b * A_ + a] - mx);
  const float inv = 1.0f / den;
  for (int a = 0; a < A_; ++a)
    pol[b * A_ + a] = (mask[b * A_ + a] != 0) ? __expf(logits[b * A_ + a] - mx) * inv : 0.0f;
}

// ================= host-side helpers =================
static inline void gemm(const float* X, const float* W, const float* b, const float* res,
                        float* Y, int M, int N, int K, int act, int rowmap, hipStream_t s)
{
  dim3 grid(N / 32, M / 64);
  gemm_wmma_k<<<grid, 32, 0, s>>>(X, W, b, res, Y, M, N, K, act, rowmap);
}
static inline int egrid(size_t n) { return (int)((n + 255) / 256); }
static inline int wgrid(int waves) { return (waves + 3) / 4; }   // 128 thr = 4 waves

extern "C" void kernel_launch(void* const* d_in, const int* in_sizes, int n_in,
                              void* d_out, int out_size, void* d_ws, size_t ws_size,
                              hipStream_t stream)
{
  (void)in_sizes; (void)n_in; (void)out_size; (void)ws_size;
  // ---- inputs (setup_inputs dict order; params dict flattened in insertion order) ----
  const float* our_team = (const float*)d_in[0];
  const float* opp_rev  = (const float*)d_in[1];
  const float* opp_pred = (const float*)d_in[2];
  const float* field    = (const float*)d_in[3];
  const int*   amask    = (const int*)d_in[4];
  const int*   omask    = (const int*)d_in[5];
  const int*   aidx     = (const int*)d_in[6];
  const int*   tera_u   = (const int*)d_in[7];
  const float* pproj_w  = (const float*)d_in[8];
  const float* pproj_b  = (const float*)d_in[9];
  const float* sa_wi    = (const float*)d_in[10];
  const float* sa_bi    = (const float*)d_in[11];
  const float* sa_wo    = (const float*)d_in[12];
  const float* sa_bo    = (const float*)d_in[13];
  const float* ca_wi    = (const float*)d_in[14];
  const float* ca_bi    = (const float*)d_in[15];
  const float* ca_wo    = (const float*)d_in[16];
  const float* ca_bo    = (const float*)d_in[17];
  const float* fproj_w  = (const float*)d_in[18];
  const float* fproj_b  = (const float*)d_in[19];
  const float* tera_tab = (const float*)d_in[20];
  const float* comb_w   = (const float*)d_in[21];
  const float* comb_b   = (const float*)d_in[22];
  const float* comb_lg  = (const float*)d_in[23];
  const float* comb_lb  = (const float*)d_in[24];
  const float* pol_w1   = (const float*)d_in[25];
  const float* pol_b1   = (const float*)d_in[26];
  const float* pol_w2   = (const float*)d_in[27];
  const float* pol_b2   = (const float*)d_in[28];
  const float* val_w1   = (const float*)d_in[29];
  const float* val_b1   = (const float*)d_in[30];
  const float* val_w2   = (const float*)d_in[31];
  const float* val_b2   = (const float*)d_in[32];
  const float* ter_w1   = (const float*)d_in[33];
  const float* ter_b1   = (const float*)d_in[34];
  const float* ter_w2   = (const float*)d_in[35];
  const float* ter_b2   = (const float*)d_in[36];
  const float* L[2][12];
  for (int l = 0; l < 2; ++l)
    for (int j = 0; j < 12; ++j)
      L[l][j] = (const float*)d_in[37 + l * 12 + j];
  // L[l]: 0 wi, 1 bi, 2 wo, 3 bo, 4 w1, 5 b1, 6 w2, 7 b2, 8 ln1_g, 9 ln1_b, 10 ln2_g, 11 ln2_b

  float* out_f = (float*)d_out;
  const float scale = 0.0883883476483184f;  // 1/sqrt(128)

  // ---- workspace carve ----
  char* wsp = (char*)d_ws;
  size_t off = 0;
  auto carve = [&](size_t nf) -> float* {
    float* p = (float*)(wsp + off);
    off += ((nf * sizeof(float) + 255) & ~(size_t)255);
    return p;
  };
  float* bufA   = carve((size_t)B12_ * 4 * H_);  // qkv / ffn / kv scratch
  float* xbuf   = carve((size_t)B12_ * H_);      // encoder x [B,12,H]
  float* buf2   = carve((size_t)B12_ * H_);      // LN out / attn ctx
  float* oppH   = carve((size_t)BT_ * H_);       // opp_h
  float* qa     = carve((size_t)B_ * H_);        // gathered active rows
  float* qp     = carve((size_t)B_ * H_);        // cross q proj / comb pre-LN
  float* cctx   = carve((size_t)B_ * H_);        // cross attn ctx / head hidden
  float* crossO = carve((size_t)B_ * H_);        // cross out (kept for feats)
  float* grep   = carve((size_t)B_ * H_);        // global repr
  float* fieldH = carve((size_t)B_ * H_);
  float* featsB = carve((size_t)B_ * (3 * H_ + H_ / 4));
  float* hid    = carve((size_t)B_ * H_);

  // 1) blend opp inputs in P-space (linear => equivalent to blending projections)
  blend_k<<<egrid((size_t)BT_ * P_), 256, 0, stream>>>(opp_rev, opp_pred, omask, bufA);
  // 2) opp_h = pproj(blend)
  gemm(bufA, pproj_w, pproj_b, nullptr, oppH, BT_, H_, P_, 0, 0, stream);
  // 3) our_h = pproj(our_team)
  gemm(our_team, pproj_w, pproj_b, nullptr, buf2, BT_, H_, P_, 0, 0, stream);
  // 4) self-MHA qkv
  gemm(buf2, sa_wi, sa_bi, nullptr, bufA, BT_, 3 * H_, H_, 0, 0, stream);
  // 5) self attention (S=6)
  attn_self_k<<<wgrid(B_ * NH_ * T_), 128, 0, stream>>>(bufA, buf2, B_, T_, scale);
  // 6) our_att = out-proj, scattered directly into x rows [b*12 + t]
  gemm(buf2, sa_wo, sa_bo, nullptr, xbuf, BT_, H_, H_, 0, 1, stream);
  copy_opp_k<<<egrid((size_t)BT_ * H_), 256, 0, stream>>>(oppH, xbuf);
  // 7) gather active rows (idx < 6 -> our_att half of x)
  gather_k<<<egrid((size_t)B_ * H_), 256, 0, stream>>>(xbuf, aidx, qa);
  // 8) cross q proj (ca_wi rows [0,H))
  gemm(qa, ca_wi, ca_bi, nullptr, qp, B_, H_, H_, 0, 0, stream);
  // 9) cross k||v proj (ca_wi rows [H,3H))
  gemm(oppH, ca_wi + (size_t)H_ * H_, ca_bi + H_, nullptr, bufA, BT_, 2 * H_, H_, 0, 0, stream);
  // 10) cross attention
  attn_cross_k<<<wgrid(B_ * NH_), 128, 0, stream>>>(qp, bufA, cctx, B_, T_, scale);
  // 11) cross out proj
  gemm(cctx, ca_wo, ca_bo, nullptr, crossO, B_, H_, H_, 0, 0, stream);

  // 12) encoder layers (norm_first)
  for (int l = 0; l < 2; ++l) {
    layernorm_k<<<wgrid(B12_), 128, 0, stream>>>(xbuf, L[l][8], L[l][9], buf2, B12_);
    gemm(buf2, L[l][0], L[l][1], nullptr, bufA, B12_, 3 * H_, H_, 0, 0, stream);
    attn_self_k<<<wgrid(B_ * NH_ * T2_), 128, 0, stream>>>(bufA, buf2, B_, T2_, scale);
    gemm(buf2, L[l][2], L[l][3], xbuf, xbuf, B12_, H_, H_, 0, 0, stream);          // x += mha
    layernorm_k<<<wgrid(B12_), 128, 0, stream>>>(xbuf, L[l][10], L[l][11], buf2, B12_);
    gemm(buf2, L[l][4], L[l][5], nullptr, bufA, B12_, 4 * H_, H_, 1, 0, stream);   // relu FFN1
    gemm(bufA, L[l][6], L[l][7], xbuf, xbuf, B12_, H_, 4 * H_, 0, 0, stream);      // x += FFN2
  }

  // 13) global repr + field proj + feats
  mean12_k<<<egrid((size_t)B_ * H_), 256, 0, stream>>>(xbuf, grep);
  gemm(field, fproj_w, fproj_b, nullptr, fieldH, B_, H_, F_, 0, 0, stream);
  feats_k<<<egrid((size_t)B_ * (3 * H_ + H_ / 4)), 256, 0, stream>>>(
      grep, crossO, fieldH, tera_tab, tera_u, featsB);

  // 14) comb: relu(lin) then LN
  gemm(featsB, comb_w, comb_b, nullptr, qp, B_, H_, 3 * H_ + H_ / 4, 1, 0, stream);
  layernorm_k<<<wgrid(B_), 128, 0, stream>>>(qp, comb_lg, comb_lb, hid, B_);

  // 15) heads  (d_out = [logits | policy | value | tera_adv])
  float* logits = out_f;
  float* policy = out_f + (size_t)B_ * A_;
  float* value  = out_f + 2 * (size_t)B_ * A_;
  float* terao  = value + B_;

  gemm(hid, pol_w1, pol_b1, nullptr, cctx, B_, H_, H_, 1, 0, stream);
  head_k<<<wgrid(B_ * A_), 128, 0, stream>>>(cctx, pol_w2, pol_b2, logits, B_, A_, H_);
  policy_k<<<egrid(B_), 256, 0, stream>>>(logits, amask, policy);

  gemm(hid, val_w1, val_b1, nullptr, cctx, B_, H_, H_, 1, 0, stream);
  head_k<<<wgrid(B_), 128, 0, stream>>>(cctx, val_w2, val_b2, value, B_, 1, H_);

  gemm(hid, ter_w1, ter_b1, nullptr, cctx, B_, H_ / 2, H_, 1, 0, stream);
  head_k<<<wgrid(B_), 128, 0, stream>>>(cctx, ter_w2, ter_b2, terao, B_, 1, H_ / 2);
}
